// LLM_62130996904446
// MI455X (gfx1250) — matrix operand order repaired and checked by reading, not compiled, not run on hardware
//
#include <hip/hip_runtime.h>
#include <hip/hip_bf16.h>
#include <math.h>

#define LYR 8
#define CDIM 1024
#define NH 16
#define HDIM 64
#define HID 4096
#define VOC 32000
#define BB 2
#define TT 1024
#define MTOK (BB * TT) // 2048 tokens

typedef __attribute__((ext_vector_type(16))) __bf16 v16bf;
typedef __attribute__((ext_vector_type(8)))  __bf16 v8bf;
typedef __attribute__((ext_vector_type(4)))  __bf16 v4bf;
typedef __attribute__((ext_vector_type(8)))  float  v8f;
typedef __attribute__((ext_vector_type(4)))  float  v4f;

// ---------- helpers ----------

__device__ __forceinline__ __bf16 b16(float f) { return (__bf16)f; }

__device__ __forceinline__ v4bf cvt4(v4f t) {
  v4bf o; o[0] = b16(t[0]); o[1] = b16(t[1]); o[2] = b16(t[2]); o[3] = b16(t[3]);
  return o;
}

__device__ __forceinline__ v16bf concat8(v8bf lo, v8bf hi) {
  return __builtin_shufflevector(lo, hi, 0, 1, 2, 3, 4, 5, 6, 7, 8, 9, 10, 11,
                                 12, 13, 14, 15);
}

__device__ __forceinline__ v8f wmma_bf16(v16bf a, v16bf b, v8f c) {
  return __builtin_amdgcn_wmma_f32_16x16x32_bf16(false, a, false, b, (short)0, c,
                                                 false, false);
}

// A fragment 16x32 bf16 from row-major [m][k] source (stride ld elements).
// lane(row=l&15, half=l>>4) holds k runs [8h,8h+8) and [16+8h,16+8h+8)
// -> two contiguous 16B LDS loads.
__device__ __forceinline__ v16bf load_a_frag(const __bf16* s, int ld) {
  const int lane = threadIdx.x & 31;
  const __bf16* base = s + (lane & 15) * ld + 8 * (lane >> 4);
  v8bf lo = *(const v8bf*)(base);
  v8bf hi = *(const v8bf*)(base + 16);
  return concat8(lo, hi);
}

// B fragment 32x16 bf16 where source is stored transposed [n][k] (stride ld).
// lane(n=l&15, half=l>>4) holds k runs [8h,8h+8) and [16+8h,16+8h+8) -> two 16B LDS loads.
__device__ __forceinline__ v16bf load_bT_frag(const __bf16* s, int ld) {
  const int lane = threadIdx.x & 31;
  const __bf16* base = s + (lane & 15) * ld + 8 * (lane >> 4);
  v8bf lo = *(const v8bf*)(base);
  v8bf hi = *(const v8bf*)(base + 16);
  return concat8(lo, hi);
}

// ---------- embedding gather ----------

__global__ __launch_bounds__(256) void k_embed(const int* __restrict__ ids,
                                               const float* __restrict__ Wemb,
                                               float* __restrict__ H) {
  const int row = blockIdx.x;
  const int id  = ids[row];
  for (int c = threadIdx.x; c < CDIM; c += 256)
    H[(size_t)row * CDIM + c] = Wemb[(size_t)id * CDIM + c];
}

// ---------- bf16 WMMA GEMM:  C = Res + alpha * (A @ B[^T]) ----------
// A: [M,K] f32. B: transB ? [N,K] : [K,N] f32. C/Res: [M,N] f32.
// Tile 64x128, 8 waves (4x2), each wave 16x64 (4 WMMA accs), K-step 32.
// Register-staged, LDS double-buffered pipeline: global loads of tile k+1
// overlap the WMMAs of tile k; one barrier per k-step.
// M % 64 == 0, N % 128 == 0, K % 32 == 0.

__global__ __launch_bounds__(256) void k_gemm(const float* __restrict__ A,
                                              const float* __restrict__ B,
                                              const float* Res, float* C,
                                              int M, int N, int K, int transB,
                                              float alpha) {
  __shared__ __bf16 As[2][64 * 32];   // [m][k]
  __shared__ __bf16 BsT[2][128 * 32]; // [n][k]
  const int tid  = threadIdx.x;
  const int bm   = blockIdx.y * 64;
  const int bn   = blockIdx.x * 128;
  const int wave = tid >> 5;
  const int lane = tid & 31;
  const int wm   = (wave >> 1) * 16;
  const int wn   = (wave & 1) * 64;

  // per-thread staging coordinates (constant across k-steps)
  const int ar = tid >> 3, ac = (tid & 7) * 4; // A: rows ar, ar+32 / col ac
  const int bk = wave;                         // B(!transB): k rows wave+8j
  const int bn4 = lane * 4;                    //            n cols bn4..bn4+3

  v4f ra0, ra1, rb[4];

  auto gload = [&](int k0) {
    ra0 = *(const v4f*)(A + (size_t)(bm + ar) * K + k0 + ac);
    ra1 = *(const v4f*)(A + (size_t)(bm + ar + 32) * K + k0 + ac);
    if (transB) {
#pragma unroll
      for (int j = 0; j < 4; ++j)
        rb[j] = *(const v4f*)(B + (size_t)(bn + ar + j * 32) * K + k0 + ac);
    } else {
#pragma unroll
      for (int j = 0; j < 4; ++j)
        rb[j] = *(const v4f*)(B + (size_t)(k0 + bk + j * 8) * N + bn + bn4);
    }
  };

  auto lstore = [&](int buf) {
    *(v4bf*)(&As[buf][ar * 32 + ac])        = cvt4(ra0);
    *(v4bf*)(&As[buf][(ar + 32) * 32 + ac]) = cvt4(ra1);
    if (transB) {
#pragma unroll
      for (int j = 0; j < 4; ++j)
        *(v4bf*)(&BsT[buf][(ar + j * 32) * 32 + ac]) = cvt4(rb[j]);
    } else {
#pragma unroll
      for (int j = 0; j < 4; ++j) {
        int k = bk + j * 8;
        BsT[buf][(bn4 + 0) * 32 + k] = b16(rb[j][0]);
        BsT[buf][(bn4 + 1) * 32 + k] = b16(rb[j][1]);
        BsT[buf][(bn4 + 2) * 32 + k] = b16(rb[j][2]);
        BsT[buf][(bn4 + 3) * 32 + k] = b16(rb[j][3]);
      }
    }
  };

  v8f acc0 = {}, acc1 = {}, acc2 = {}, acc3 = {};

  const int nk = K >> 5;
  gload(0);
  for (int kt = 0; kt < nk; ++kt) {
    const int buf = kt & 1;
    lstore(buf);
    __syncthreads();
    if (kt + 1 < nk) {
      gload((kt + 1) * 32); // in-flight during this tile's WMMAs
      if (kt + 2 < nk) {    // L2 prefetch two tiles ahead
        const int kp = (kt + 2) * 32;
        if (tid < 64) __builtin_prefetch(&A[(size_t)(bm + tid) * K + kp], 0, 3);
        if (transB) {
          if (tid >= 64 && tid < 192)
            __builtin_prefetch(&B[(size_t)(bn + tid - 64) * K + kp], 0, 3);
        } else {
          if (tid >= 64 && tid < 96)
            __builtin_prefetch(&B[(size_t)(kp + tid - 64) * N + bn], 0, 3);
        }
      }
    }
    const __bf16* as = As[buf];
    const __bf16* bs = BsT[buf];
    v16bf a = load_a_frag(as + wm * 32, 32);
    acc0 = wmma_bf16(a, load_bT_frag(bs + (wn + 0)  * 32, 32), acc0);
    acc1 = wmma_bf16(a, load_bT_frag(bs + (wn + 16) * 32, 32), acc1);
    acc2 = wmma_bf16(a, load_bT_frag(bs + (wn + 32) * 32, 32), acc2);
    acc3 = wmma_bf16(a, load_bT_frag(bs + (wn + 48) * 32, 32), acc3);
  }

  const int half = lane >> 4;
  const int cn   = lane & 15;
#pragma unroll
  for (int v = 0; v < 8; ++v) {
    size_t rowb = (size_t)(bm + wm + v + 8 * half) * N + bn + wn + cn;
    float a0 = alpha * acc0[v], a1 = alpha * acc1[v];
    float a2 = alpha * acc2[v], a3 = alpha * acc3[v];
    if (Res) {
      C[rowb + 0]  = Res[rowb + 0]  + a0;
      C[rowb + 16] = Res[rowb + 16] + a1;
      C[rowb + 32] = Res[rowb + 32] + a2;
      C[rowb + 48] = Res[rowb + 48] + a3;
    } else {
      C[rowb + 0] = a0; C[rowb + 16] = a1; C[rowb + 32] = a2; C[rowb + 48] = a3;
    }
  }
}

// ---------- fused causal attention (flash-style online softmax) ----------
// One wave per (b, h, 16-query tile). Streams 32-key tiles; QK^T and PV in WMMA.

__global__ __launch_bounds__(32) void k_attn(const float* __restrict__ Q,
                                             const float* __restrict__ Kb,
                                             const float* __restrict__ Vb,
                                             float* __restrict__ O) {
  __shared__ __bf16 qs[16 * 64];  // [q][d]
  __shared__ __bf16 ks[32 * 64];  // [key][d]  (QK^T B-frags read contiguous d)
  __shared__ __bf16 vsT[64 * 32]; // [d][key]  (PV B-frags read contiguous key)
  __shared__ float  ss[16 * 32];
  __shared__ __bf16 ps[16 * 32];
  __shared__ float  rowm[16], rowl[16], fsc[16];

  const int tile = blockIdx.x & 63;        // T/16 tiles
  const int h    = (blockIdx.x >> 6) & 15;
  const int b    = blockIdx.x >> 10;
  const int q0   = tile * 16;
  const int lane = threadIdx.x;
  const int half = lane >> 4;
  const int ln   = lane & 15;

  // stage Q tile (256 float4)
#pragma unroll
  for (int i = lane; i < 256; i += 32) {
    int r = i >> 4, c = (i & 15) * 4;
    v4f t = *(const v4f*)(Q + (size_t)(b * TT + q0 + r) * CDIM + h * HDIM + c);
    *(v4bf*)(qs + r * 64 + c) = cvt4(t);
  }
  if (lane < 16) { rowm[lane] = -1e30f; rowl[lane] = 0.f; }
  __syncthreads();

  const v16bf aQ0 = load_a_frag(qs, 64);      // head dims 0..31
  const v16bf aQ1 = load_a_frag(qs + 32, 64); // head dims 32..63
  v8f o0 = {}, o1 = {}, o2 = {}, o3 = {};

  const int nkt = (q0 + 16 + 31) / 32; // causal window in 32-key tiles
  for (int kt = 0; kt < nkt; ++kt) {
    const int k0 = kt * 32;
#pragma unroll
    for (int i = lane; i < 512; i += 32) { // 32x64 floats = 512 float4
      int    r  = i >> 4, c = (i & 15) * 4;
      size_t gi = (size_t)(b * TT + k0 + r) * CDIM + h * HDIM + c;
      v4f tk = *(const v4f*)(Kb + gi);
      v4f tv = *(const v4f*)(Vb + gi);
      *(v4bf*)(ks + r * 64 + c) = cvt4(tk);
      vsT[(c + 0) * 32 + r] = b16(tv[0]);
      vsT[(c + 1) * 32 + r] = b16(tv[1]);
      vsT[(c + 2) * 32 + r] = b16(tv[2]);
      vsT[(c + 3) * 32 + r] = b16(tv[3]);
    }
    __syncthreads();

    // S = Q K^T : two 16-key groups x two 32-dim K-steps
#pragma unroll
    for (int nn = 0; nn < 2; ++nn) {
      v8f s = {};
      s = wmma_bf16(aQ0, load_bT_frag(ks + nn * 16 * 64, 64), s);
      s = wmma_bf16(aQ1, load_bT_frag(ks + nn * 16 * 64 + 32, 64), s);
#pragma unroll
      for (int v = 0; v < 8; ++v) {
        int r = v + 8 * half;
        int c = nn * 16 + ln;
        ss[r * 32 + c] = ((k0 + c) <= (q0 + r)) ? s[v] * 0.125f /*1/sqrt(64)*/
                                                : -1e30f;
      }
    }
    __syncthreads();

    if (lane < 16) { // per-row online softmax update
      float mx = -1e30f;
#pragma unroll
      for (int j = 0; j < 32; ++j) mx = fmaxf(mx, ss[lane * 32 + j]);
      float mold = rowm[lane];
      float mnew = fmaxf(mold, mx);
      float f    = __expf(mold - mnew);
      float psum = 0.f;
#pragma unroll
      for (int j = 0; j < 32; ++j) {
        float p = __expf(ss[lane * 32 + j] - mnew);
        ps[lane * 32 + j] = b16(p);
        psum += p;
      }
      rowl[lane] = rowl[lane] * f + psum;
      rowm[lane] = mnew;
      fsc[lane]  = f;
    }
    __syncthreads();

#pragma unroll
    for (int v = 0; v < 8; ++v) { // rescale running output
      float f = fsc[v + 8 * half];
      o0[v] *= f; o1[v] *= f; o2[v] *= f; o3[v] *= f;
    }
    const v16bf aP = load_a_frag(ps, 32); // probs 16x32
    o0 = wmma_bf16(aP, load_bT_frag(vsT + 0  * 32, 32), o0);
    o1 = wmma_bf16(aP, load_bT_frag(vsT + 16 * 32, 32), o1);
    o2 = wmma_bf16(aP, load_bT_frag(vsT + 32 * 32, 32), o2);
    o3 = wmma_bf16(aP, load_bT_frag(vsT + 48 * 32, 32), o3);
    __syncthreads();
  }

#pragma unroll
  for (int v = 0; v < 8; ++v) {
    int    r    = v + 8 * half;
    float  inv  = 1.f / rowl[r];
    size_t base = (size_t)(b * TT + q0 + r) * CDIM + h * HDIM;
    O[base + 0  + ln] = o0[v] * inv;
    O[base + 16 + ln] = o1[v] * inv;
    O[base + 32 + ln] = o2[v] * inv;
    O[base + 48 + ln] = o3[v] * inv;
  }
}

// ---------- SwiGLU pointwise:  mid = fc * silu(gate) / sqrt(HID) ----------

__global__ __launch_bounds__(256) void k_swiglu(const float* __restrict__ FC,
                                                const float* __restrict__ G,
                                                float* __restrict__ Mid,
                                                size_t n) {
  size_t i = (size_t)blockIdx.x * 256 + threadIdx.x;
  if (i < n) {
    float g = G[i];
    Mid[i]  = FC[i] * (g / (1.f + __expf(-g))) * 0.015625f; // 1/sqrt(4096)
  }
}

// ---------- cross entropy ----------

__global__ void k_zero2(float* a) { a[0] = 0.f; a[1] = 0.f; }

__global__ __launch_bounds__(256) void k_xent(const float* __restrict__ logits,
                                              const int* __restrict__ labels,
                                              float* accum) {
  __shared__ float red[256];
  const int    row = blockIdx.x;
  const int    tid = threadIdx.x;
  const float* lr  = logits + (size_t)row * VOC;

  float mx = -1e30f;
  for (int j = tid; j < VOC; j += 256) mx = fmaxf(mx, lr[j]);
  red[tid] = mx;
  __syncthreads();
  for (int s = 128; s > 0; s >>= 1) {
    if (tid < s) red[tid] = fmaxf(red[tid], red[tid + s]);
    __syncthreads();
  }
  mx = red[0];
  __syncthreads();

  float sum = 0.f;
  for (int j = tid; j < VOC; j += 256) sum += __expf(lr[j] - mx);
  red[tid] = sum;
  __syncthreads();
  for (int s = 128; s > 0; s >>= 1) {
    if (tid < s) red[tid] += red[tid + s];
    __syncthreads();
  }
  if (tid == 0) {
    int lab = labels[row];
    if (lab != -1) {
      float lp = lr[lab] - mx - __logf(red[0]);
      atomicAdd(&accum[0], -lp);
      atomicAdd(&accum[1], 1.f);
    }
  }
}

__global__ void k_finloss(const float* a, float* out) {
  out[0] = a[0] / fmaxf(a[1], 1.f);
}

// ---------- driver ----------

extern "C" void kernel_launch(void* const* d_in, const int* in_sizes, int n_in,
                              void* d_out, int out_size, void* d_ws,
                              size_t ws_size, hipStream_t stream) {
  (void)in_sizes; (void)n_in; (void)out_size; (void)ws_size;

  const int*   ids    = (const int*)d_in[0];
  const int*   labels = (const int*)d_in[1];
  const float* Wemb   = (const float*)d_in[2];
  const float* Wq     = (const float*)d_in[3];
  const float* Wk     = (const float*)d_in[4];
  const float* Wv     = (const float*)d_in[5];
  const float* Wo     = (const float*)d_in[6];
  const float* Wfc    = (const float*)d_in[7];
  const float* Wgate  = (const float*)d_in[8];
  const float* Wproj  = (const float*)d_in[9];
  float*       out    = (float*)d_out; // [B*T*V logits] + [1 loss]

  float*       ws = (float*)d_ws;
  const size_t S1 = (size_t)MTOK * CDIM;
  const size_t S2 = (size_t)MTOK * HID;
  float* h     = ws;
  float* q     = h + S1;
  float* kbuf  = q + S1;
  float* vbuf  = kbuf + S1;
  float* attn  = vbuf + S1;
  float* fc    = attn + S1;
  float* gate  = fc + S2;
  float* mid   = gate + S2;
  float* accum = mid + S2;

  const float invL = 0.35355339059327373f; // 1/sqrt(8 layers)

  k_embed<<<MTOK, 256, 0, stream>>>(ids, Wemb, h);

  dim3 gD(CDIM / 128, MTOK / 64); // 8 x 32
  dim3 gH(HID / 128, MTOK / 64);  // 32 x 32
  dim3 gV(VOC / 128, MTOK / 64);  // 250 x 32

  for (int l = 0; l < LYR; ++l) {
    const float* wq = Wq    + (size_t)l * CDIM * CDIM;
    const float* wk = Wk    + (size_t)l * CDIM * CDIM;
    const float* wv = Wv    + (size_t)l * CDIM * CDIM;
    const float* wo = Wo    + (size_t)l * CDIM * CDIM;
    const float* wf = Wfc   + (size_t)l * CDIM * HID;
    const float* wg = Wgate + (size_t)l * CDIM * HID;
    const float* wp = Wproj + (size_t)l * HID * CDIM;

    k_gemm<<<gD, 256, 0, stream>>>(h, wq, nullptr, q,    MTOK, CDIM, CDIM, 0, 1.f);
    k_gemm<<<gD, 256, 0, stream>>>(h, wk, nullptr, kbuf, MTOK, CDIM, CDIM, 0, 1.f);
    k_gemm<<<gD, 256, 0, stream>>>(h, wv, nullptr, vbuf, MTOK, CDIM, CDIM, 0, 1.f);

    k_attn<<<BB * NH * (TT / 16), 32, 0, stream>>>(q, kbuf, vbuf, attn);

    // h = h + (attn @ Wo) / sqrt(L)
    k_gemm<<<gD, 256, 0, stream>>>(attn, wo, h, h, MTOK, CDIM, CDIM, 0, invL);

    k_gemm<<<gH, 256, 0, stream>>>(h, wf, nullptr, fc,   MTOK, HID, CDIM, 0, 1.f);
    k_gemm<<<gH, 256, 0, stream>>>(h, wg, nullptr, gate, MTOK, HID, CDIM, 0, 1.f);

    const size_t nmid = (size_t)MTOK * HID;
    k_swiglu<<<(unsigned)((nmid + 255) / 256), 256, 0, stream>>>(fc, gate, mid, nmid);

    // h = h + (mid @ Wproj) / sqrt(L)
    k_gemm<<<gD, 256, 0, stream>>>(mid, wp, h, h, MTOK, CDIM, HID, 0, invL);
  }

  // logits = h @ Wemb^T -> d_out
  k_gemm<<<gV, 256, 0, stream>>>(h, Wemb, nullptr, out, MTOK, VOC, CDIM, 1, 1.f);

  k_zero2<<<1, 1, 0, stream>>>(accum);
  k_xent<<<MTOK, 256, 0, stream>>>(out, labels, accum);
  k_finloss<<<1, 1, 0, stream>>>(accum, out + (size_t)MTOK * VOC);
}
